// SpectralModel_31911607009658
// MI455X (gfx1250) — matrix operand order repaired and checked by reading, not compiled, not run on hardware
//
#include <hip/hip_runtime.h>
#include <hip/hip_bf16.h>

// ---------------- problem constants ----------------
#define Nn    8
#define CD    64
#define Hh    128
#define Ww    128
#define Ee    128
#define NHh   4
#define KS    7
#define Pp    49
#define DFFv  512
#define NLv   4
#define Mtot  (Nn*Hh*Ww)     // 131072 pixels
#define HP    (Hh+2)         // conv pad=1
#define WP    (Ww+2)
#define HA    (Hh+6)         // attn pad=3
#define WA    (Ww+6)

typedef _Float16 h16;
typedef __attribute__((ext_vector_type(16))) _Float16 v16h;
typedef __attribute__((ext_vector_type(8)))  _Float16 v8h;
typedef __attribute__((ext_vector_type(4)))  _Float16 v4h;
typedef __attribute__((ext_vector_type(8)))  float    v8f;

// ---------------- WMMA helpers ----------------
__device__ inline v8f vzero8() {
  v8f z;
#pragma unroll
  for (int i = 0; i < 8; ++i) z[i] = 0.0f;
  return z;
}

__device__ inline v16h frag_join(v8h lo, v8h hi) {
  v16h r;
#pragma unroll
  for (int i = 0; i < 8; ++i) { r[i] = lo[i]; r[8 + i] = hi[i]; }
  return r;
}

// A fragment (16x32 f16, MxK): lane l holds row m=l&15, K-half kh=(l>>4)*8:
//   a[0..7]  = A[m][kh+0..7]      a[8..15] = A[m][16+kh+0..7]
__device__ inline v16h load_a_frag(const h16* Abase, int lda) {
  int lane = threadIdx.x & 31;
  const h16* p = Abase + (long)(lane & 15) * lda + ((lane >> 4) << 3);
  return frag_join(*(const v8h*)p, *(const v8h*)(p + 16));
}

// B fragment from pre-packed layout: block of 32 lanes x 16 f16, contiguous.
__device__ inline v16h load_b_frag(const h16* Bblk) {
  int lane = threadIdx.x & 31;
  const h16* p = Bblk + lane * 16;
  return frag_join(*(const v8h*)p, *(const v8h*)(p + 8));
}

__device__ inline v8f wmma_f16(v16h a, v16h b, v8f c) {
  return __builtin_amdgcn_wmma_f32_16x16x32_f16(
      false, a, false, b, (short)0, c, false, false);
}

__device__ inline float wave_sum32(float x) {
#pragma unroll
  for (int m = 1; m < 32; m <<= 1) x += __shfl_xor(x, m, 32);
  return x;
}

// ---------------- weight packing ----------------
// dst laid out as [kt][nt][lane][i] (i=0..15), with source element
//   B(k,n) = src[off + k*sk + n*sn],  k = kt*32 + (i>>3)*16 + (lane>>4)*8 + (i&7),
//   n = nt*16 + (lane&15)
__global__ void pack_b_kernel(h16* dst, const float* src, int K, int N,
                              int sk, int sn, long off) {
  int tid = blockIdx.x * blockDim.x + threadIdx.x;
  int total = K * N;
  if (tid >= total) return;
  int i    = tid & 15;
  int lane = (tid >> 4) & 31;
  int tile = tid >> 9;
  int NT   = N >> 4;
  int nt   = tile % NT;
  int kt   = tile / NT;
  int n = nt * 16 + (lane & 15);
  int k = kt * 32 + ((i >> 3) << 4) + ((lane >> 4) << 3) + (i & 7);
  dst[tid] = (h16)src[off + (long)k * sk + (long)n * sn];
}

// ---------------- NCHW f32 -> padded NHWC f16 ----------------
__global__ void nhwc_pad_kernel(h16* X0p, const float* dct) {
  int pix = blockIdx.x * blockDim.x + threadIdx.x;
  if (pix >= Mtot) return;
  int w = pix & (Ww - 1);
  int h = (pix >> 7) & (Hh - 1);
  int n = pix >> 14;
  const float* s = dct + (long)n * CD * Hh * Ww + (long)h * Ww + w;
  h16* d = X0p + (((long)n * HP + (h + 1)) * WP + (w + 1)) * CD;
#pragma unroll 4
  for (int c = 0; c < CD; ++c) d[c] = (h16)s[(long)c * Hh * Ww];
}

// ---------------- conv3x3 + FiLM (9 shifted WMMA GEMMs) ----------------
__global__ __launch_bounds__(256) void conv_film_kernel(
    h16* __restrict__ T, const h16* __restrict__ X0p,
    const h16* __restrict__ c1p, const float* __restrict__ c1b,
    const float* __restrict__ qt) {
  __shared__ __align__(16) h16 stg[8][16 * 64];
  int wave = threadIdx.x >> 5, lane = threadIdx.x & 31;
  long mt = (long)blockIdx.x * 8 + wave;
  long m0 = mt * 16;
  int w0 = (int)(m0 & (Ww - 1));
  int h0 = (int)((m0 >> 7) & (Hh - 1));
  int ni = (int)(m0 >> 14);

  v8f acc[4];
#pragma unroll
  for (int t = 0; t < 4; ++t) acc[t] = vzero8();

  for (int tap = 0; tap < 9; ++tap) {
    int dy = tap / 3, dx = tap - 3 * dy;
    const h16* Abase = X0p + (((long)ni * HP + (h0 + dy)) * WP + (w0 + dx)) * CD;
    const h16* Bt = c1p + tap * (64 * 64);
#pragma unroll
    for (int kt = 0; kt < 2; ++kt) {
      v16h a = load_a_frag(Abase + kt * 32, CD);
#pragma unroll
      for (int nt = 0; nt < 4; ++nt) {
        v16h b = load_b_frag(Bt + (long)(kt * 4 + nt) * 512);
        acc[nt] = wmma_f16(a, b, acc[nt]);
      }
    }
  }
  // t = dct + qt * (conv + b1)
  const h16* Dbase = X0p + (((long)ni * HP + (h0 + 1)) * WP + (w0 + 1)) * CD;
  int rbase = (lane >> 4) * 8, col0 = lane & 15;
#pragma unroll
  for (int nt = 0; nt < 4; ++nt) {
    int c = nt * 16 + col0;
    float qv = qt[ni * CD + c];
    float bv = c1b[c];
#pragma unroll
    for (int r = 0; r < 8; ++r) {
      int m = rbase + r;
      float dctv = (float)Dbase[(long)m * CD + c];
      stg[wave][m * 64 + c] = (h16)(dctv + qv * (acc[nt][r] + bv));
    }
  }
  __syncthreads();
  const v8h* sp = (const v8h*)stg[wave];
  v8h* dp = (v8h*)(T + m0 * 64);
  for (int j = lane; j < 128; j += 32) dp[j] = sp[j];
}

// ---------------- generic GEMM: out(M x 128) = A(M x K) @ Bp + bias ----------------
// padded!=0 : rows addressed into (Nn,HA,WA,Ee) buffer at interior offset (+3,+3)
__global__ __launch_bounds__(256) void gemm128_kernel(
    h16* __restrict__ out, int padded,
    const h16* __restrict__ A, int K,
    const h16* __restrict__ Bp, const float* __restrict__ bias) {
  __shared__ __align__(16) h16 stg[8][16 * 128];
  int wave = threadIdx.x >> 5, lane = threadIdx.x & 31;
  long mt = (long)blockIdx.x * 8 + wave;
  long m0 = mt * 16;

  v8f acc[8];
#pragma unroll
  for (int t = 0; t < 8; ++t) acc[t] = vzero8();

  const h16* Arow = A + m0 * K;
  int KT = K >> 5;
  for (int kt = 0; kt < KT; ++kt) {
    v16h a = load_a_frag(Arow + kt * 32, K);
#pragma unroll
    for (int nt = 0; nt < 8; ++nt) {
      v16h b = load_b_frag(Bp + (long)(kt * 8 + nt) * 512);
      acc[nt] = wmma_f16(a, b, acc[nt]);
    }
  }
  int rbase = (lane >> 4) * 8, col0 = lane & 15;
#pragma unroll
  for (int nt = 0; nt < 8; ++nt) {
    int c = nt * 16 + col0;
    float bv = bias[c];
#pragma unroll
    for (int r = 0; r < 8; ++r)
      stg[wave][(rbase + r) * 128 + c] = (h16)(acc[nt][r] + bv);
  }
  __syncthreads();
  h16* dst;
  if (padded) {
    int w0 = (int)(m0 & (Ww - 1));
    int h0 = (int)((m0 >> 7) & (Hh - 1));
    int ni = (int)(m0 >> 14);
    dst = out + (((long)ni * HA + (h0 + 3)) * WA + (w0 + 3)) * Ee;
  } else {
    dst = out + m0 * Ee;
  }
  const v8h* sp = (const v8h*)stg[wave];
  v8h* dp = (v8h*)dst;
  for (int j = lane; j < 256; j += 32) dp[j] = sp[j];
}

// ---------------- LayerNorm (optionally of X + X2) ----------------
__global__ __launch_bounds__(256) void ln_kernel(
    h16* __restrict__ Y, const h16* __restrict__ X, const h16* __restrict__ X2,
    const float* __restrict__ g, const float* __restrict__ b) {
  int wave = threadIdx.x >> 5, lane = threadIdx.x & 31;
  long pix = (long)blockIdx.x * 8 + wave;
  int c0 = lane * 4;
  const h16* x = X + pix * Ee + c0;
  float v[4];
#pragma unroll
  for (int i = 0; i < 4; ++i) v[i] = (float)x[i];
  if (X2) {
    const h16* x2 = X2 + pix * Ee + c0;
#pragma unroll
    for (int i = 0; i < 4; ++i) v[i] += (float)x2[i];
  }
  float s = v[0] + v[1] + v[2] + v[3];
  float mean = wave_sum32(s) * (1.0f / Ee);
  float sq = 0.f;
#pragma unroll
  for (int i = 0; i < 4; ++i) { v[i] -= mean; sq += v[i] * v[i]; }
  float var = wave_sum32(sq) * (1.0f / Ee);
  float inv = rsqrtf(var + 1e-5f);
  h16* y = Y + pix * Ee + c0;
#pragma unroll
  for (int i = 0; i < 4; ++i)
    y[i] = (h16)(v[i] * inv * g[c0 + i] + b[c0 + i]);
}

// ---------------- 7x7 local attention, online softmax ----------------
__global__ __launch_bounds__(256) void attn_kernel(
    h16* __restrict__ O, const h16* __restrict__ Q,
    const h16* __restrict__ Kp, const h16* __restrict__ Vp,
    const float* __restrict__ rb) {
  int wave = threadIdx.x >> 5, lane = threadIdx.x & 31;
  long pix = (long)blockIdx.x * 8 + wave;
  int w = (int)(pix & (Ww - 1));
  int h = (int)((pix >> 7) & (Hh - 1));
  int ni = (int)(pix >> 14);
  int head = lane >> 3;           // 8 lanes per head, dk=32
  int c0 = lane * 4;

  const h16* qp = Q + pix * Ee + c0;
  const float scale = 0.08838834764831845f;   // 1/sqrt(128)
  float q0 = (float)qp[0] * scale, q1 = (float)qp[1] * scale;
  float q2 = (float)qp[2] * scale, q3 = (float)qp[3] * scale;

  float mmax = -1e30f, lsum = 0.f;
  float o0 = 0.f, o1 = 0.f, o2 = 0.f, o3 = 0.f;
  const float* rbh = rb + head * Pp;

  for (int p = 0; p < Pp; ++p) {
    int dy = p / KS, dx = p - dy * KS;
    long base = (((long)ni * HA + (h + dy)) * WA + (w + dx)) * Ee + c0;
    const h16* kp = Kp + base;
    float s = q0 * (float)kp[0] + q1 * (float)kp[1] +
              q2 * (float)kp[2] + q3 * (float)kp[3];
    s += __shfl_xor(s, 1, 32);
    s += __shfl_xor(s, 2, 32);
    s += __shfl_xor(s, 4, 32);
    s += rbh[p];
    float mn = fmaxf(mmax, s);
    float corr = __expf(mmax - mn);
    float e = __expf(s - mn);
    const h16* vp = Vp + base;
    o0 = o0 * corr + e * (float)vp[0];
    o1 = o1 * corr + e * (float)vp[1];
    o2 = o2 * corr + e * (float)vp[2];
    o3 = o3 * corr + e * (float)vp[3];
    lsum = lsum * corr + e;
    mmax = mn;
  }
  float inv = 1.0f / lsum;
  v4h r;
  r[0] = (h16)(o0 * inv); r[1] = (h16)(o1 * inv);
  r[2] = (h16)(o2 * inv); r[3] = (h16)(o3 * inv);
  *(v4h*)(O + pix * Ee + c0) = r;
}

// ---------------- fused FFN: X = resid + gelu(Y@W1+b1)@W2 + b2 ----------------
__global__ __launch_bounds__(128) void ffn_kernel(
    h16* __restrict__ Xout, const h16* __restrict__ Y,
    const h16* __restrict__ Resid,
    const h16* __restrict__ B1p, const float* __restrict__ b1,
    const h16* __restrict__ B2p, const float* __restrict__ b2) {
  __shared__ __align__(16) h16 hbuf[4][16 * DFFv];   // 64 KB
  int wave = threadIdx.x >> 5, lane = threadIdx.x & 31;
  long mt = (long)blockIdx.x * 4 + wave;
  long m0 = mt * 16;
  const h16* Arow = Y + m0 * Ee;
  int rbase = (lane >> 4) * 8, col0 = lane & 15;

  // phase 1: hidden = gelu(Y@W1 + b1) -> LDS (16 x 512)
  for (int ntg = 0; ntg < 32; ntg += 8) {
    v8f acc[8];
#pragma unroll
    for (int t = 0; t < 8; ++t) acc[t] = vzero8();
#pragma unroll
    for (int kt = 0; kt < 4; ++kt) {
      v16h a = load_a_frag(Arow + kt * 32, Ee);
#pragma unroll
      for (int j = 0; j < 8; ++j) {
        v16h b = load_b_frag(B1p + (long)(kt * 32 + ntg + j) * 512);
        acc[j] = wmma_f16(a, b, acc[j]);
      }
    }
#pragma unroll
    for (int j = 0; j < 8; ++j) {
      int c = (ntg + j) * 16 + col0;
      float bv = b1[c];
#pragma unroll
      for (int r = 0; r < 8; ++r) {
        float x = acc[j][r] + bv;
        float gelu = 0.5f * x * (1.0f + erff(x * 0.70710678118654752f));
        hbuf[wave][(rbase + r) * DFFv + c] = (h16)gelu;
      }
    }
  }
  __syncthreads();

  // phase 2: out = hidden@W2 + b2 + resid
  v8f acc2[8];
#pragma unroll
  for (int t = 0; t < 8; ++t) acc2[t] = vzero8();
  const h16* Hrow = hbuf[wave];
  for (int kt = 0; kt < 16; ++kt) {
    v16h a = load_a_frag(Hrow + kt * 32, DFFv);
#pragma unroll
    for (int nt = 0; nt < 8; ++nt) {
      v16h b = load_b_frag(B2p + (long)(kt * 8 + nt) * 512);
      acc2[nt] = wmma_f16(a, b, acc2[nt]);
    }
  }
  __syncthreads();

  const h16* Rrow = Resid + m0 * Ee;
#pragma unroll
  for (int nt = 0; nt < 8; ++nt) {
    int c = nt * 16 + col0;
    float bv = b2[c];
#pragma unroll
    for (int r = 0; r < 8; ++r) {
      int m = rbase + r;
      float val = acc2[nt][r] + bv + (float)Rrow[(long)m * Ee + c];
      hbuf[wave][m * Ee + c] = (h16)val;   // reuse LDS as store staging
    }
  }
  __syncthreads();
  const v8h* sp = (const v8h*)hbuf[wave];
  v8h* dp = (v8h*)(Xout + m0 * Ee);
  for (int j = lane; j < 256; j += 32) dp[j] = sp[j];
}

// ---------------- decoder: out = dct + (X@Wd^T + bd) * qt  (fp32 NCHW) --------
__global__ __launch_bounds__(256) void decoder_kernel(
    float* __restrict__ out, const float* __restrict__ dct,
    const h16* __restrict__ X, const h16* __restrict__ Bp,
    const float* __restrict__ db, const float* __restrict__ qt) {
  int wave = threadIdx.x >> 5, lane = threadIdx.x & 31;
  long mt = (long)blockIdx.x * 8 + wave;
  long m0 = mt * 16;
  int w0 = (int)(m0 & (Ww - 1));
  int h0 = (int)((m0 >> 7) & (Hh - 1));
  int ni = (int)(m0 >> 14);

  v8f acc[4];
#pragma unroll
  for (int t = 0; t < 4; ++t) acc[t] = vzero8();
  const h16* Arow = X + m0 * Ee;
#pragma unroll
  for (int kt = 0; kt < 4; ++kt) {
    v16h a = load_a_frag(Arow + kt * 32, Ee);
#pragma unroll
    for (int nt = 0; nt < 4; ++nt) {
      v16h b = load_b_frag(Bp + (long)(kt * 4 + nt) * 512);
      acc[nt] = wmma_f16(a, b, acc[nt]);
    }
  }
  int rbase = (lane >> 4) * 8, col0 = lane & 15;
#pragma unroll
  for (int nt = 0; nt < 4; ++nt) {
    int c = nt * 16 + col0;
    float qv = qt[ni * CD + c];
    float bv = db[c];
    long base = (((long)ni * CD + c) * Hh + h0) * Ww + w0;
#pragma unroll
    for (int r = 0; r < 8; ++r) {
      int m = rbase + r;
      out[base + m] = dct[base + m] + (acc[nt][r] + bv) * qv;
    }
  }
}

// ---------------- host orchestration ----------------
static inline size_t alignup256(size_t x) { return (x + 255) & ~(size_t)255; }

extern "C" void kernel_launch(void* const* d_in, const int* in_sizes, int n_in,
                              void* d_out, int out_size, void* d_ws, size_t ws_size,
                              hipStream_t stream) {
  (void)in_sizes; (void)n_in; (void)out_size; (void)ws_size;
  const float* dct    = (const float*)d_in[0];
  const float* qt     = (const float*)d_in[1];
  const float* c1w    = (const float*)d_in[2];
  const float* c1b    = (const float*)d_in[3];
  const float* c2w    = (const float*)d_in[4];
  const float* c2b    = (const float*)d_in[5];
  const float* ln1g   = (const float*)d_in[6];
  const float* ln1b   = (const float*)d_in[7];
  const float* qw     = (const float*)d_in[8];
  const float* qb     = (const float*)d_in[9];
  const float* kw     = (const float*)d_in[10];
  const float* kb     = (const float*)d_in[11];
  const float* vw     = (const float*)d_in[12];
  const float* vb     = (const float*)d_in[13];
  const float* relb   = (const float*)d_in[14];
  const float* ln2g   = (const float*)d_in[15];
  const float* ln2b   = (const float*)d_in[16];
  const float* fw1    = (const float*)d_in[17];
  const float* fb1    = (const float*)d_in[18];
  const float* fw2    = (const float*)d_in[19];
  const float* fb2    = (const float*)d_in[20];
  const float* dw     = (const float*)d_in[21];
  const float* db     = (const float*)d_in[22];
  float* outp = (float*)d_out;

  char* w = (char*)d_ws;
  auto take = [&](size_t nelem) -> h16* {
    h16* p = (h16*)w;
    w += alignup256(nelem * sizeof(h16));
    return p;
  };
  const size_t szX0p = (size_t)Nn * HP * WP * CD;       // 8,652,800
  const size_t szT   = (size_t)Mtot * CD;
  const size_t szX   = (size_t)Mtot * Ee;
  const size_t szKp  = (size_t)Nn * HA * WA * Ee;       // 18,386,944
  h16* X0p = take(szX0p);
  h16* T   = take(szT);
  h16* X   = take(szX);
  h16* Y   = take(szX);
  h16* Q   = take(szX);
  h16* Kp  = take(szKp);
  h16* Vp  = take(szKp);
  h16* A2  = take(szX);
  h16* c1p  = take(9 * 64 * 64);
  h16* c2p  = take(64 * 128);
  h16* qkvp = take((size_t)NLv * 3 * 128 * 128);
  h16* f1p  = take((size_t)NLv * 128 * 512);
  h16* f2p  = take((size_t)NLv * 512 * 128);
  h16* decp = take(128 * 64);

  hipMemsetAsync(X0p, 0, szX0p * sizeof(h16), stream);
  hipMemsetAsync(Kp,  0, szKp  * sizeof(h16), stream);
  hipMemsetAsync(Vp,  0, szKp  * sizeof(h16), stream);

  auto pack = [&](h16* dst, const float* src, int K, int N, int sk, int sn, long off) {
    int total = K * N;
    pack_b_kernel<<<(total + 255) / 256, 256, 0, stream>>>(dst, src, K, N, sk, sn, off);
  };
  // conv taps: B(k=cin, n=cout) = c1w[cout*576 + cin*9 + tap]
  for (int tap = 0; tap < 9; ++tap)
    pack(c1p + tap * 4096, c1w, 64, 64, 9, 576, tap);
  // embed: (E,CD) out,in -> B(k,n) = c2w[n*64 + k]
  pack(c2p, c2w, 64, 128, 1, 64, 0);
  for (int l = 0; l < NLv; ++l) {
    pack(qkvp + (size_t)(l * 3 + 0) * 16384, qw, 128, 128, 128, 1, (long)l * 16384);
    pack(qkvp + (size_t)(l * 3 + 1) * 16384, kw, 128, 128, 128, 1, (long)l * 16384);
    pack(qkvp + (size_t)(l * 3 + 2) * 16384, vw, 128, 128, 128, 1, (long)l * 16384);
    pack(f1p + (size_t)l * 65536, fw1, 128, 512, 512, 1, (long)l * 65536);
    pack(f2p + (size_t)l * 65536, fw2, 512, 128, 128, 1, (long)l * 65536);
  }
  // decoder: (CD,E) out,in -> B(k,n) = dw[n*128 + k]
  pack(decp, dw, 128, 64, 1, 128, 0);

  // encoder
  nhwc_pad_kernel<<<Mtot / 256, 256, 0, stream>>>(X0p, dct);
  conv_film_kernel<<<Mtot / 128, 256, 0, stream>>>(T, X0p, c1p, c1b, qt);
  gemm128_kernel<<<Mtot / 128, 256, 0, stream>>>(X, 0, T, 64, c2p, c2b);

  // transformer layers
  for (int l = 0; l < NLv; ++l) {
    ln_kernel<<<Mtot / 8, 256, 0, stream>>>(Y, X, nullptr, ln1g + l * Ee, ln1b + l * Ee);
    gemm128_kernel<<<Mtot / 128, 256, 0, stream>>>(Q,  0, Y, 128, qkvp + (size_t)(l * 3 + 0) * 16384, qb + l * Ee);
    gemm128_kernel<<<Mtot / 128, 256, 0, stream>>>(Kp, 1, Y, 128, qkvp + (size_t)(l * 3 + 1) * 16384, kb + l * Ee);
    gemm128_kernel<<<Mtot / 128, 256, 0, stream>>>(Vp, 1, Y, 128, qkvp + (size_t)(l * 3 + 2) * 16384, vb + l * Ee);
    attn_kernel<<<Mtot / 8, 256, 0, stream>>>(A2, Q, Kp, Vp, relb + l * NHh * Pp);
    ln_kernel<<<Mtot / 8, 256, 0, stream>>>(Y, X, A2, ln2g + l * Ee, ln2b + l * Ee);
    ffn_kernel<<<Mtot / 64, 128, 0, stream>>>(X, Y, A2,
        f1p + (size_t)l * 65536, fb1 + l * DFFv,
        f2p + (size_t)l * 65536, fb2 + l * Ee);
  }

  // decoder
  decoder_kernel<<<Mtot / 128, 256, 0, stream>>>(outp, dct, X, decp, db, qt);
}
